// CasualAttention_79216376808249
// MI455X (gfx1250) — matrix operand order repaired and checked by reading, not compiled, not run on hardware
//
#include <hip/hip_runtime.h>

// Causal attention forward (flash-attention style) for gfx1250 / MI455X.
// B=2, H=16, S=2048, head_dim=64, fp32 in/out, bf16 WMMA compute.
// One block = one (b,h) x 128 query rows (8 waves x 16-row tiles).
// 64-key chunks fetched once per block via the Tensor Data Mover
// (double-buffered), converted cooperatively to bf16, consumed by all waves.
// Causal masking is unswitched; wave-level control is scalarized via
// readfirstlane so WMMA regions keep EXEC all-ones (ISA 7.12 requirement).

#define DM 1024   // d_model
#define NH 16     // heads
#define SQ 2048   // sequence
#define BT 2      // batch
#define HD 64     // head dim
#define QT 16     // query rows per wave tile
#define KC 64     // keys per chunk
#define WPB 8     // waves per block
#define QROWS (WPB * QT)   // 128 query rows per block

typedef __attribute__((ext_vector_type(16))) __bf16 v16bf;
typedef __attribute__((ext_vector_type(8)))  float  v8f;
typedef __attribute__((ext_vector_type(4)))  unsigned v4u;
typedef __attribute__((ext_vector_type(8)))  int      v8i;
typedef __attribute__((ext_vector_type(4)))  int      v4i;

union BF16x16 { v16bf v; unsigned u[8]; };

__device__ __forceinline__ unsigned pack_bf16(float a, float b) {
  union { float f; unsigned u; } ca, cb;
  ca.f = a; cb.f = b;
  unsigned ra = (ca.u + 0x7FFFu + ((ca.u >> 16) & 1u)) >> 16;
  unsigned rb = (cb.u + 0x7FFFu + ((cb.u >> 16) & 1u)) & 0xFFFF0000u;
  return (ra & 0xFFFFu) | rb;
}

__device__ __forceinline__ unsigned short f2bf(float a) {
  union { float f; unsigned u; } c; c.f = a;
  return (unsigned short)((c.u + 0x7FFFu + ((c.u >> 16) & 1u)) >> 16);
}

// Issue a TDM 2-D tile load: KC rows x HD f32, row stride DM f32, into LDS.
// D# layout per CDNA5 ISA ch.8 (group0: 4 dw, group1: 8 dw, groups 2/3 unused).
__device__ __forceinline__ void tdm_load_tile(const float* gsrc, unsigned ldsByteOff) {
  const unsigned long long ga = (unsigned long long)(const void*)gsrc;
  v4u g0;
  g0[0] = 1u;                                                // count=1, user D#
  g0[1] = ldsByteOff;                                        // lds_addr (bytes)
  g0[2] = (unsigned)(ga & 0xFFFFFFFFu);                      // global_addr[31:0]
  g0[3] = (unsigned)((ga >> 32) & 0x01FFFFFFu) | (2u << 30); // addr[56:32]|type=2
  v8i g1;
  g1[0] = (int)(2u << 16);            // wg_mask=0 | data_size=2 (4B) | no flags
  g1[1] = (int)((unsigned)HD << 16);  // tensor_dim0 = 64
  g1[2] = (int)((unsigned)KC << 16);  // tensor_dim1 = 64
  g1[3] = (int)((unsigned)HD << 16);  // tile_dim0 = 64
  g1[4] = (int)((unsigned)KC);        // tile_dim1 = 64 | tile_dim2 = 0
  g1[5] = (int)((unsigned)DM);        // tensor_dim0_stride = 1024 elements
  g1[6] = 0;
  g1[7] = 0;
  const v4i z4 = {0, 0, 0, 0};
  const v8i z8 = {0, 0, 0, 0, 0, 0, 0, 0};
  __builtin_amdgcn_tensor_load_to_lds(g0, g1, z4, z4, z8, 0);
}

__device__ __forceinline__ unsigned lds_offset(const void* p) {
  // Flat LDS pointers carry the byte offset in the low 32 bits (ISA 10.2).
  return (unsigned)(unsigned long long)p;
}

__global__ __launch_bounds__(WPB * 32, 1)
void fa_fwd(const float* __restrict__ x, float* __restrict__ out) {
  const int lane = threadIdx.x & 31;
  // Wave index as a compiler-known scalar: all control flow derived from it
  // becomes SALU + s_cbranch (no exec masking around WMMA).
  const int wv = __builtin_amdgcn_readfirstlane((int)(threadIdx.x >> 5));

  const int blocksPerBH = SQ / QROWS;            // 16
  const int bh = blockIdx.x / blocksPerBH;
  const int qb = (blockIdx.x - bh * blocksPerBH) * QROWS;
  const int b  = bh / NH;
  const int h  = bh - b * NH;
  const int qBase = qb + wv * QT;                // scalar (SGPR)

  const float* xb = x   + (size_t)b * SQ * DM + (size_t)h * HD;
  float*       ob = out + (size_t)b * SQ * DM + (size_t)h * HD;

  // ---- LDS ----
  __shared__ __align__(16) float          stg[2][KC * HD];       // TDM dst, 32 KB
  __shared__ __align__(16) unsigned short kT[HD * KC];           // K^T [d][key], 8 KB
  __shared__ __align__(16) unsigned short vB[KC * HD];           // V [key][d], 8 KB
  __shared__ __align__(16) unsigned short qlds[WPB][QT * HD];    // 16 KB
  __shared__ __align__(16) unsigned short plds[WPB][QT * KC];    // 16 KB

  const int m16 = lane & 15;
  const int hf  = lane >> 4;

  // ---- stage this wave's Q tile to LDS as bf16 ----
  {
    const float* qrow = xb + (size_t)(qBase + m16) * DM + hf * 32;
    unsigned* qd = (unsigned*)(&qlds[wv][0] + m16 * 64 + hf * 32);
#pragma unroll
    for (int i = 0; i < 8; ++i) {
      const float4 f = ((const float4*)qrow)[i];
      qd[i * 2 + 0] = pack_bf16(f.x, f.y);
      qd[i * 2 + 1] = pack_bf16(f.z, f.w);
    }
  }
  __builtin_amdgcn_wave_barrier();

  // ---- Q A-fragments: qa0 d=0..31, qa1 d=32..63 ----
  BF16x16 qa0, qa1;
  {
    const unsigned* ql32 = (const unsigned*)&qlds[wv][0];
#pragma unroll
    for (int p = 0; p < 8; ++p) {
      const int k0 = ((p >> 2) << 4) + hf * 8 + ((p & 3) << 1);
      qa0.u[p] = ql32[m16 * 32 + (k0 >> 1)];
      qa1.u[p] = ql32[m16 * 32 + 16 + (k0 >> 1)];
    }
  }

  v8f o0 = {}, o1 = {}, o2 = {}, o3 = {};
  float mr[8], lr[8];
#pragma unroll
  for (int v = 0; v < 8; ++v) { mr[v] = -__builtin_inff(); lr[v] = 0.f; }

  const int myChunks    = (qBase >> 6) + 1;      // scalar
  const int blockChunks = ((qb + 112) >> 6) + 1; // scalar
  const float cexp = 0.125f * 1.44269504088896340736f;  // 1/sqrt(64)*log2(e)

  const unsigned* kTu = (const unsigned*)kT;
  const unsigned* vBu = (const unsigned*)vB;

  // S B-fragment loader: column tile ct (16 keys), d-half dh (32 dims)
  auto load_kfrag = [&](int ct, int dh, BF16x16& kb) {
    const uint4* kp = (const uint4*)(vBu + (ct * 16 + m16) * 32 + dh * 16 + hf * 4);
    const uint4 a = kp[0], c = kp[2];
    kb.u[0] = a.x; kb.u[1] = a.y; kb.u[2] = a.z; kb.u[3] = a.w;
    kb.u[4] = c.x; kb.u[5] = c.y; kb.u[6] = c.z; kb.u[7] = c.w;
  };
  auto s_tile = [&](int ct) -> v8f {
    BF16x16 kb0, kb1;
    load_kfrag(ct, 0, kb0);
    load_kfrag(ct, 1, kb1);
    v8f acc = {};
    acc = __builtin_amdgcn_wmma_f32_16x16x32_bf16(false, qa0.v, false, kb0.v,
                                                  (short)0, acc, false, false);
    acc = __builtin_amdgcn_wmma_f32_16x16x32_bf16(false, qa1.v, false, kb1.v,
                                                  (short)0, acc, false, false);
    return acc;
  };
  // PV for one 32-key half hh
  auto pv_half = [&](int hh) {
    BF16x16 pa;
    const unsigned* pl32 = (const unsigned*)&plds[wv][0];
#pragma unroll
    for (int p = 0; p < 8; ++p) {
      const int k0 = ((p >> 2) << 4) + hf * 8 + ((p & 3) << 1);
      pa.u[p] = pl32[m16 * 32 + hh * 16 + (k0 >> 1)];
    }
#pragma unroll
    for (int t = 0; t < 4; ++t) {
      BF16x16 vb;
      const uint4* vp = (const uint4*)(kTu + (t * 16 + m16) * 32 + hh * 16 + hf * 4);
      const uint4 w0 = vp[0], w1 = vp[2];
      vb.u[0] = w0.x; vb.u[1] = w0.y; vb.u[2] = w0.z; vb.u[3] = w0.w;
      vb.u[4] = w1.x; vb.u[5] = w1.y; vb.u[6] = w1.z; vb.u[7] = w1.w;
      v8f& ot = (t == 0) ? o0 : (t == 1) ? o1 : (t == 2) ? o2 : o3;
      ot = __builtin_amdgcn_wmma_f32_16x16x32_bf16(false, pa.v, false, vb.v,
                                                   (short)0, ot, false, false);
    }
  };

  // ---- prologue: DMA chunk 0 into buffer 0 ----
  if (wv == 0) tdm_load_tile(xb, lds_offset(&stg[0][0]));

  for (int j = 0; j < blockChunks; ++j) {
    const int jBase = j * KC;

    if (wv == 0) __builtin_amdgcn_s_wait_tensorcnt(0);
    __syncthreads();                              // TDM tile visible to block

    // ---- cooperative f32 -> bf16 convert: vB [key][d], kT [d][key] ----
    {
      const float* sb = &stg[j & 1][0];
#pragma unroll
      for (int i = 0; i < 8; ++i) {
        const int base = i * 512 + (int)threadIdx.x * 2;   // 4096 elems / 256 thr
        const float2 f = *(const float2*)(sb + base);
        const unsigned u = pack_bf16(f.x, f.y);
        ((unsigned*)vB)[base >> 1] = u;
        const int key = base >> 6, d = base & 63;
        kT[d * 64 + key]       = (unsigned short)(u & 0xFFFFu);
        kT[(d + 1) * 64 + key] = (unsigned short)(u >> 16);
      }
    }

    // overlap: issue next chunk's DMA into the other buffer now
    if (wv == 0 && (j + 1) < blockChunks)
      tdm_load_tile(xb + (size_t)(jBase + KC) * DM, lds_offset(&stg[(j + 1) & 1][0]));

    __syncthreads();                              // convert done

    if (j < myChunks - 1) {
      // ================= interior chunk: no causal masking =================
      v8f s0 = s_tile(0), s1 = s_tile(1), s2 = s_tile(2), s3 = s_tile(3);
#pragma unroll
      for (int v = 0; v < 8; ++v) {
        const float x0 = s0[v], x1 = s1[v], x2 = s2[v], x3 = s3[v];
        float t = fmaxf(fmaxf(x0, x1), fmaxf(x2, x3));
        t = fmaxf(t, __shfl_xor(t, 1, 32));
        t = fmaxf(t, __shfl_xor(t, 2, 32));
        t = fmaxf(t, __shfl_xor(t, 4, 32));
        t = fmaxf(t, __shfl_xor(t, 8, 32));
        const float mnew  = fmaxf(mr[v], t);
        const float alpha = __builtin_amdgcn_exp2f((mr[v] - mnew) * cexp);
        mr[v] = mnew;
        const float p0 = __builtin_amdgcn_exp2f((x0 - mnew) * cexp);
        const float p1 = __builtin_amdgcn_exp2f((x1 - mnew) * cexp);
        const float p2 = __builtin_amdgcn_exp2f((x2 - mnew) * cexp);
        const float p3 = __builtin_amdgcn_exp2f((x3 - mnew) * cexp);
        float rs = (p0 + p1) + (p2 + p3);
        rs += __shfl_xor(rs, 1, 32);
        rs += __shfl_xor(rs, 2, 32);
        rs += __shfl_xor(rs, 4, 32);
        rs += __shfl_xor(rs, 8, 32);
        lr[v] = lr[v] * alpha + rs;
        o0[v] *= alpha; o1[v] *= alpha; o2[v] *= alpha; o3[v] *= alpha;
        const int rl = v + (hf << 3);
        plds[wv][rl * 64 +      m16] = f2bf(p0);
        plds[wv][rl * 64 + 16 + m16] = f2bf(p1);
        plds[wv][rl * 64 + 32 + m16] = f2bf(p2);
        plds[wv][rl * 64 + 48 + m16] = f2bf(p3);
      }
      __builtin_amdgcn_wave_barrier();
      pv_half(0);
      pv_half(1);
    } else if (j == myChunks - 1) {
      // ================= diagonal chunk: masked path =======================
      const int off  = qBase & 63;        // 0,16,32,48 (scalar)
      const int diag = off >> 4;          // tile index of the diagonal (scalar)
      v8f s0 = {}, s1 = {}, s2 = {}, s3 = {};
      s0 = s_tile(0);
      if (diag >= 1) s1 = s_tile(1);
      if (diag >= 2) s2 = s_tile(2);
      if (diag >= 3) s3 = s_tile(3);
      const int rowg = qBase + (hf << 3);
#pragma unroll
      for (int v = 0; v < 8; ++v) {
        const int row = rowg + v;
        const float NI = -__builtin_inff();
        float xv[4];
        xv[0] = (diag > 0) ? s0[v] : ((jBase + m16      <= row) ? s0[v] : NI);
        xv[1] = (diag < 1) ? NI : (diag > 1) ? s1[v]
                : ((jBase + 16 + m16 <= row) ? s1[v] : NI);
        xv[2] = (diag < 2) ? NI : (diag > 2) ? s2[v]
                : ((jBase + 32 + m16 <= row) ? s2[v] : NI);
        xv[3] = (diag < 3) ? NI
                : ((jBase + 48 + m16 <= row) ? s3[v] : NI);
        float t = fmaxf(fmaxf(xv[0], xv[1]), fmaxf(xv[2], xv[3]));
        t = fmaxf(t, __shfl_xor(t, 1, 32));
        t = fmaxf(t, __shfl_xor(t, 2, 32));
        t = fmaxf(t, __shfl_xor(t, 4, 32));
        t = fmaxf(t, __shfl_xor(t, 8, 32));
        const float mnew  = fmaxf(mr[v], t);
        const float alpha = __builtin_amdgcn_exp2f((mr[v] - mnew) * cexp);
        mr[v] = mnew;
        const float p0 = __builtin_amdgcn_exp2f((xv[0] - mnew) * cexp);
        const float p1 = __builtin_amdgcn_exp2f((xv[1] - mnew) * cexp);
        const float p2 = __builtin_amdgcn_exp2f((xv[2] - mnew) * cexp);
        const float p3 = __builtin_amdgcn_exp2f((xv[3] - mnew) * cexp);
        float rs = (p0 + p1) + (p2 + p3);
        rs += __shfl_xor(rs, 1, 32);
        rs += __shfl_xor(rs, 2, 32);
        rs += __shfl_xor(rs, 4, 32);
        rs += __shfl_xor(rs, 8, 32);
        lr[v] = lr[v] * alpha + rs;
        o0[v] *= alpha; o1[v] *= alpha; o2[v] *= alpha; o3[v] *= alpha;
        const int rl = v + (hf << 3);
        plds[wv][rl * 64 +      m16] = f2bf(p0);
        plds[wv][rl * 64 + 16 + m16] = f2bf(p1);
        plds[wv][rl * 64 + 32 + m16] = f2bf(p2);
        plds[wv][rl * 64 + 48 + m16] = f2bf(p3);
      }
      __builtin_amdgcn_wave_barrier();
      pv_half(0);
      if (diag >= 2) pv_half(1);   // half 1 is entirely masked otherwise
    }

    __syncthreads();                              // kT/vB reusable next chunk
  }

  // ---- normalize and store (f32) ----
#pragma unroll
  for (int v = 0; v < 8; ++v) {
    const int row = qBase + v + (hf << 3);
    const float inv = 1.0f / lr[v];
    float* orow = ob + (size_t)row * DM;
    orow[ 0 + m16] = o0[v] * inv;
    orow[16 + m16] = o1[v] * inv;
    orow[32 + m16] = o2[v] * inv;
    orow[48 + m16] = o3[v] * inv;
  }
}

extern "C" void kernel_launch(void* const* d_in, const int* in_sizes, int n_in,
                              void* d_out, int out_size, void* d_ws, size_t ws_size,
                              hipStream_t stream) {
  (void)in_sizes; (void)n_in; (void)out_size; (void)d_ws; (void)ws_size;
  const float* x = (const float*)d_in[0];
  float* out = (float*)d_out;
  const int blocks = BT * NH * (SQ / QROWS);   // 512
  dim3 grid(blocks);
  dim3 block(WPB * 32);                        // 256 threads = 8 waves
  hipLaunchKernelGGL(fa_fwd, grid, block, 0, stream, x, out);
}
